// HardBinaryConv_43490838839974
// MI455X (gfx1250) — 8-bit matrix operand order repaired and checked by reading, compiled, not run on hardware
//
#include <hip/hip_runtime.h>


typedef __attribute__((ext_vector_type(16))) int v16i;
typedef __attribute__((ext_vector_type(8)))  float v8f;

#define NB      32
#define IN_CHN  256
#define OUT_CHN 256
#define HH      56
#define WW      56

#define FP8_P1  ((unsigned char)0x38)
#define FP8_M1  ((unsigned char)0xB8)

__global__ void binarize_x_kernel(const float* __restrict__ x,
                                  unsigned char* __restrict__ bx) {
    int idx = blockIdx.x * blockDim.x + threadIdx.x;
    int w = idx % WW;
    int h = (idx / WW) % HH;
    int c = (idx / (WW * HH)) % IN_CHN;
    int n = idx / (WW * HH * IN_CHN);
    float v = x[idx];
    bx[(((n * HH + h) * WW + w) * IN_CHN) + c] = (v >= 0.0f) ? FP8_P1 : FP8_M1;
}

__global__ void pack_w_kernel(const float* __restrict__ wflat,
                              unsigned char* __restrict__ wpk) {
    int idx = blockIdx.x * blockDim.x + threadIdx.x;
    int kw = idx % 3;
    int kh = (idx / 3) % 3;
    int c  = (idx / 9) % IN_CHN;
    int o  = idx / (9 * IN_CHN);
    float v = wflat[idx];
    unsigned char s = (v >= 0.0f) ? FP8_P1 : FP8_M1;
    int kpos   = kh * 3 + kw;
    int grp    = c >> 7;
    int k      = c & 127;
    int blk64  = k >> 6;
    int k64    = k & 63;
    int p      = (k & 15) + ((k >> 5) << 4) + (((k >> 4) & 1) << 6);
    int pos    = p;
    wpk[((kpos * OUT_CHN + o) * IN_CHN) + (grp << 7) + pos] = s;
}

__global__ __launch_bounds__(256) void bconv_wmma_kernel(
        const unsigned char* __restrict__ bx,
        const unsigned char* __restrict__ wpk,
        float* __restrict__ out) {
    __shared__ unsigned char tile[3 * 18 * IN_CHN];

    int t     = blockIdx.x;
    int wtile = t & 3;
    int h     = (t >> 2) % HH;
    int n     = t / (4 * HH);
    int wbase = wtile * 16;

    for (int idx = threadIdx.x; idx < 54 * 16; idx += 256) {
        int sub = idx & 15;
        int pos = idx >> 4;
        int r   = pos / 18;
        int cc  = pos % 18;
        int hh  = h + r - 1;
        int ww  = wbase + cc - 1;
        int4 v; v.x = 0; v.y = 0; v.z = 0; v.w = 0;
        if (hh >= 0 && hh < HH && ww >= 0 && ww < WW) {
            v = *(const int4*)(bx + (((n * HH + hh) * WW + ww) * IN_CHN) + sub * 16);
        }
        *(int4*)(tile + pos * IN_CHN + sub * 16) = v;
    }
    __syncthreads();

    int wave  = threadIdx.x >> 5;
    int lane  = threadIdx.x & 31;
    int lhalf = lane >> 4;
    int lcol  = lane & 15;
    int oc0   = wave * 16;

    v8f acc0 = {0.f, 0.f, 0.f, 0.f, 0.f, 0.f, 0.f, 0.f};
    v8f acc1 = {0.f, 0.f, 0.f, 0.f, 0.f, 0.f, 0.f, 0.f};

    for (int kpos = 0; kpos < 9; ++kpos) {
        int kh = kpos / 3;
        int kw = kpos % 3;
        const unsigned char* wrow0 =
            wpk + ((size_t)(kpos * OUT_CHN + oc0 + lcol)) * IN_CHN + lhalf * 64;
        const unsigned char* wrow1 = wrow0 + (size_t)128 * IN_CHN;
        const unsigned char* lrow =
            tile + (kh * 18 + kw + lcol) * IN_CHN + lhalf * 16;

        __builtin_prefetch(wrow0 + OUT_CHN * IN_CHN, 0, 1);
        __builtin_prefetch(wrow1 + OUT_CHN * IN_CHN, 0, 1);

#pragma unroll
        for (int cch = 0; cch < 2; ++cch) {
            v16i a0 = *(const v16i*)(wrow0 + cch * 128);
            v16i a1 = *(const v16i*)(wrow1 + cch * 128);
            union { v16i v; int4 q[4]; } bu;
            bu.q[0] = *(const int4*)(lrow + cch * 128);
            bu.q[1] = *(const int4*)(lrow + cch * 128 + 32);
            bu.q[2] = *(const int4*)(lrow + cch * 128 + 64);
            bu.q[3] = *(const int4*)(lrow + cch * 128 + 96);
            acc0 = __builtin_amdgcn_wmma_f32_16x16x128_fp8_fp8(
                       a0, bu.v, (short)0, acc0, false, false);
            acc1 = __builtin_amdgcn_wmma_f32_16x16x128_fp8_fp8(
                       a1, bu.v, (short)0, acc1, false, false);
        }
    }

    int wcol = wbase + lcol;
    if (wcol < WW) {
        size_t base = (((size_t)n * OUT_CHN) * HH + h) * WW + wcol;
#pragma unroll
        for (int r = 0; r < 8; ++r) {
            int oc = oc0 + r + 8 * lhalf;
            out[base + (size_t)oc * HH * WW]         = acc0[r];
            out[base + (size_t)(oc + 128) * HH * WW] = acc1[r];
        }
    }
}

extern "C" void kernel_launch(void* const* d_in, const int* in_sizes, int n_in,
                              void* d_out, int out_size, void* d_ws, size_t ws_size,
                              hipStream_t stream) {
    const float* x     = (const float*)d_in[0];
    const float* wflat = (const float*)d_in[1];
    float*       out   = (float*)d_out;

    unsigned char* bx  = (unsigned char*)d_ws;
    unsigned char* wpk = bx + (size_t)NB * HH * WW * IN_CHN;

    const int x_elems = NB * IN_CHN * HH * WW;
    const int w_elems = OUT_CHN * IN_CHN * 9;

    binarize_x_kernel<<<x_elems / 256, 256, 0, stream>>>(x, bx);
    pack_w_kernel<<<w_elems / 256, 256, 0, stream>>>(wflat, wpk);

    dim3 grid(NB * HH * 4);
    bconv_wmma_kernel<<<grid, 256, 0, stream>>>(bx, wpk, out);
}
